// SiameseAttentionPlus_76922864271968
// MI455X (gfx1250) — matrix operand order repaired and checked by reading, not compiled, not run on hardware
//
#include <hip/hip_runtime.h>
#include <hip/hip_bf16.h>

// ---------------------------------------------------------------------------
// SiameseAttentionPlus for MI455X (gfx1250, wave32, WMMA)
//   B=8, C=512, H=W=32, N=4 heads  ->  BN=32, Cn=128, L=1024
//
// Pipeline:
//   1) proj_kernel  : Y = (W @ X)^T in bf16, stored [L][Cn] row-major so both
//                     A-frags (rows) and B-frags (cols via transposed rows)
//                     are contiguous 16B loads matching the wave32 WMMA layouts.
//   2) attn_kernel  : per 16-row tile, stream 64 m-tiles of
//                     scores = Kt x Qt^T and value = V1t x V2t^T through
//                     v_wmma_f32_16x16x32_bf16, fusing an online softmax*value
//                     accumulation (never materializes the 1024x1024 tensors).
//   3) out_kernel   : res -> relu(w5 . res), memory-bound elementwise.
//
// Workspace (bf16 projections ~34MB) stays resident in the 192MB L2.
// ---------------------------------------------------------------------------

typedef __attribute__((ext_vector_type(16))) __bf16 v16bf;
typedef __attribute__((ext_vector_type(8)))  __bf16 v8bf;
typedef __attribute__((ext_vector_type(8)))  float  v8f;

#define L_DIM 1024
#define CN    128
#define BN_DIM 32
#define C_DIM 512
#define B_DIM 8

// ---------------------------------------------------------------------------
// Kernel 1: projection.  Y[bn][l][o] = sum_c W[o][c] * X[bn][c][l]  (bf16 out)
// One block per bn, 8 waves, weights (bf16) staged in LDS.
// ---------------------------------------------------------------------------
__global__ __launch_bounds__(256)
void proj_kernel(const float* __restrict__ x,   // [BN][CN][L]
                 const float* __restrict__ wa,  // [CN][CN]
                 const float* __restrict__ wb,  // [CN][CN]
                 __bf16* __restrict__ Ya,       // [BN][L][CN]
                 __bf16* __restrict__ Yb)       // [BN][L][CN]
{
    __shared__ __bf16 w_lds[2][CN * CN];   // 64 KB of the 320 KB WGP LDS

    const int bn   = blockIdx.x;
    const int tid  = threadIdx.x;
    const int wave = tid >> 5;
    const int lane = tid & 31;
    const int lhalf = lane >> 4;   // 0: lanes 0-15, 1: lanes 16-31
    const int lmod  = lane & 15;

    for (int i = tid; i < CN * CN; i += 256) {
        w_lds[0][i] = (__bf16)wa[i];
        w_lds[1][i] = (__bf16)wb[i];
    }
    __syncthreads();

    const float* xb = x + (size_t)bn * CN * L_DIM;

    for (int p = 0; p < 2; ++p) {
        const __bf16* W = w_lds[p];
        __bf16* Y = (p == 0 ? Ya : Yb) + (size_t)bn * L_DIM * CN;

        // each wave owns every 8th 16-wide N(=l) tile
        for (int nt = wave; nt < L_DIM / 16; nt += 8) {
            const int n0 = nt * 16;

            // B fragments: B[k=c, n=l]; lanes 0-15 hold K=0..15, 16-31 K=16..31
            v16bf bfrag[4];
#pragma unroll
            for (int kk = 0; kk < 4; ++kk) {
                const int kbase = kk * 32 + lhalf * 16;
                const int col   = n0 + lmod;
#pragma unroll
                for (int j = 0; j < 16; ++j)
                    bfrag[kk][j] = (__bf16)xb[(size_t)(kbase + j) * L_DIM + col];
            }

            for (int mt = 0; mt < CN / 16; ++mt) {
                const int o0 = mt * 16;
                v8f acc = {};
#pragma unroll
                for (int kk = 0; kk < 4; ++kk) {
                    // A frag (16x32 bf16): lane row = o0+lmod,
                    // lanes 0-15: K 0..15 ; lanes 16-31: K 16..31
                    const __bf16* arow = W + (size_t)(o0 + lmod) * CN + kk * 32 + lhalf * 16;
                    v8bf alo = *(const v8bf*)(arow);
                    v8bf ahi = *(const v8bf*)(arow + 8);
                    v16bf afrag;
#pragma unroll
                    for (int j = 0; j < 8; ++j) { afrag[j] = alo[j]; afrag[8 + j] = ahi[j]; }
                    acc = __builtin_amdgcn_wmma_f32_16x16x32_bf16(
                        false, afrag, false, bfrag[kk], (short)0, acc, false, false);
                }
                // D layout: lane col n=lmod, rows M = lhalf*8 + r  -> the 8
                // accumulator rows are contiguous in 'o': one 16B store writes
                // the transposed tile Y[l][o].
                v8bf outv;
#pragma unroll
                for (int j = 0; j < 8; ++j) outv[j] = (__bf16)acc[j];
                *(v8bf*)(Y + (size_t)(n0 + lmod) * CN + o0 + lhalf * 8) = outv;
            }
        }
    }
}

// ---------------------------------------------------------------------------
// Kernel 2: fused scores/value WMMA + online softmax*value reduction.
// grid = (BN, L/128); 8 waves/block; wave w owns rows l0 = slab*128 + w*16.
// ---------------------------------------------------------------------------
__global__ __launch_bounds__(256)
void attn_kernel(const __bf16* __restrict__ Kt,   // [BN][L][CN]
                 const __bf16* __restrict__ Qt,   // [BN][L][CN]
                 const __bf16* __restrict__ V1t,  // [BN][L][CN]
                 const __bf16* __restrict__ V2t,  // [BN][L][CN]
                 float* __restrict__ res)         // [BN][L]
{
    const int bn   = blockIdx.x;
    const int slab = blockIdx.y;
    const int tid  = threadIdx.x;
    const int wave = tid >> 5;
    const int lane = tid & 31;
    const int lhalf = lane >> 4;
    const int lmod  = lane & 15;
    const int l0 = slab * 128 + wave * 16;

    const __bf16* Ktb = Kt  + ((size_t)bn * L_DIM + l0) * CN;
    const __bf16* V1b = V1t + ((size_t)bn * L_DIM + l0) * CN;
    const __bf16* Qb  = Qt  + (size_t)bn * L_DIM * CN;
    const __bf16* V2b = V2t + (size_t)bn * L_DIM * CN;

    // Preload A fragments for this wave's 16 rows (K = 128 -> 4 chunks each)
    v16bf ak[4], av[4];
#pragma unroll
    for (int kk = 0; kk < 4; ++kk) {
        const __bf16* r1 = Ktb + (size_t)lmod * CN + kk * 32 + lhalf * 8;
        const __bf16* r2 = V1b + (size_t)lmod * CN + kk * 32 + lhalf * 8;
        v8bf a0 = *(const v8bf*)r1, a1 = *(const v8bf*)(r1 + 16);
        v8bf b0 = *(const v8bf*)r2, b1 = *(const v8bf*)(r2 + 16);
#pragma unroll
        for (int j = 0; j < 8; ++j) {
            ak[kk][j] = a0[j]; ak[kk][8 + j] = a1[j];
            av[kk][j] = b0[j]; av[kk][8 + j] = b1[j];
        }
    }

    float rmax[8], rsum[8], rwsum[8];
#pragma unroll
    for (int r = 0; r < 8; ++r) { rmax[r] = -3.4e38f; rsum[r] = 0.f; rwsum[r] = 0.f; }

    for (int mt = 0; mt < L_DIM / 16; ++mt) {
        const int m0 = mt * 16;
        const size_t rowoff = (size_t)(m0 + lmod) * CN + lhalf * 8;

        if (mt + 1 < L_DIM / 16) {   // global_prefetch_b8 for next tile
            __builtin_prefetch(Qb  + rowoff + 16 * CN, 0, 3);
            __builtin_prefetch(V2b + rowoff + 16 * CN, 0, 3);
        }

        // B fragments: lane = column m0+lmod; lanes 0-15 K=0..7|16..23, 16-31 K=8..15|24..31
        v16bf bq[4], bv[4];
#pragma unroll
        for (int kk = 0; kk < 4; ++kk) {
            const __bf16* q = Qb  + rowoff + kk * 32;
            const __bf16* v = V2b + rowoff + kk * 32;
            v8bf q0 = *(const v8bf*)q, q1 = *(const v8bf*)(q + 16);
            v8bf v0 = *(const v8bf*)v, v1 = *(const v8bf*)(v + 16);
#pragma unroll
            for (int j = 0; j < 8; ++j) {
                bq[kk][j] = q0[j]; bq[kk][8 + j] = q1[j];
                bv[kk][j] = v0[j]; bv[kk][8 + j] = v1[j];
            }
        }

        v8f sc = {}, vl = {};
#pragma unroll
        for (int kk = 0; kk < 4; ++kk) {
            sc = __builtin_amdgcn_wmma_f32_16x16x32_bf16(
                false, ak[kk], false, bq[kk], (short)0, sc, false, false);
            vl = __builtin_amdgcn_wmma_f32_16x16x32_bf16(
                false, av[kk], false, bv[kk], (short)0, vl, false, false);
        }

        // Online softmax * value: each lane accumulates the columns it sees.
#pragma unroll
        for (int r = 0; r < 8; ++r) {
            const float s = sc[r], v = vl[r];
            const float nm = fmaxf(rmax[r], s);
            const float eo = __expf(rmax[r] - nm);
            const float en = __expf(s - nm);
            rsum[r]  = rsum[r]  * eo + en;
            rwsum[r] = rwsum[r] * eo + en * v;
            rmax[r]  = nm;
        }
    }

    // Cross-lane log-sum-exp merge over the 16-lane column group.
#pragma unroll
    for (int off = 1; off < 16; off <<= 1) {
#pragma unroll
        for (int r = 0; r < 8; ++r) {
            const float om = __shfl_xor(rmax[r],  off, 32);
            const float os = __shfl_xor(rsum[r],  off, 32);
            const float ow = __shfl_xor(rwsum[r], off, 32);
            const float nm = fmaxf(rmax[r], om);
            const float s0 = __expf(rmax[r] - nm);
            const float s1 = __expf(om - nm);
            rsum[r]  = rsum[r]  * s0 + os * s1;
            rwsum[r] = rwsum[r] * s0 + ow * s1;
            rmax[r]  = nm;
        }
    }

    if (lmod == 0) {
        // lane 0 holds rows 0..7, lane 16 rows 8..15 of this tile
#pragma unroll
        for (int r = 0; r < 8; ++r)
            res[(size_t)bn * L_DIM + l0 + lhalf * 8 + r] = rwsum[r] / rsum[r];
    }
}

// ---------------------------------------------------------------------------
// Kernel 3: out[b][o][hw] = relu( sum_n w5[o][n] * res[b*4+n][hw] )
// ---------------------------------------------------------------------------
__global__ __launch_bounds__(256)
void out_kernel(const float* __restrict__ res,   // [BN][L]
                const float* __restrict__ w5,    // [C][N]
                float* __restrict__ out)         // [B][C][L]
{
    const size_t idx = (size_t)blockIdx.x * 256 + threadIdx.x;
    const int hw = (int)(idx & (L_DIM - 1));
    const int o  = (int)((idx >> 10) & (C_DIM - 1));
    const int b  = (int)(idx >> 19);
    float acc = 0.f;
#pragma unroll
    for (int n = 0; n < 4; ++n)
        acc += w5[o * 4 + n] * res[(size_t)(b * 4 + n) * L_DIM + hw];
    out[idx] = fmaxf(acc, 0.f);
}

// ---------------------------------------------------------------------------
extern "C" void kernel_launch(void* const* d_in, const int* in_sizes, int n_in,
                              void* d_out, int out_size, void* d_ws, size_t ws_size,
                              hipStream_t stream)
{
    const float* first  = (const float*)d_in[0];
    const float* second = (const float*)d_in[1];
    const float* w1 = (const float*)d_in[2];
    const float* w2 = (const float*)d_in[3];
    const float* w3 = (const float*)d_in[4];
    const float* w4 = (const float*)d_in[5];
    const float* w5 = (const float*)d_in[6];
    float* out = (float*)d_out;

    const size_t projBytes = (size_t)BN_DIM * L_DIM * CN * sizeof(__bf16);  // 8 MB
    char* ws = (char*)d_ws;
    __bf16* Kt  = (__bf16*)(ws + 0 * projBytes);
    __bf16* Qt  = (__bf16*)(ws + 1 * projBytes);
    __bf16* V1t = (__bf16*)(ws + 2 * projBytes);
    __bf16* V2t = (__bf16*)(ws + 3 * projBytes);
    float*  res = (float*)(ws + 4 * projBytes);

    proj_kernel<<<BN_DIM, 256, 0, stream>>>(first,  w1, w3, Kt,  V1t);
    proj_kernel<<<BN_DIM, 256, 0, stream>>>(second, w2, w4, Qt,  V2t);
    attn_kernel<<<dim3(BN_DIM, L_DIM / 128), 256, 0, stream>>>(Kt, Qt, V1t, V2t, res);
    out_kernel<<<(B_DIM * C_DIM * L_DIM) / 256, 256, 0, stream>>>(res, w5, out);
}
